// FiveLayerGAT_80238579024177
// MI455X (gfx1250) — matrix operand order repaired and checked by reading, not compiled, not run on hardware
//
#include <hip/hip_runtime.h>
#include <hip/hip_bf16.h>
#include <math.h>

// ---------------- problem constants (match reference) ----------------
#define N_NODES   50000
#define N_EDGES   800000
#define E_TOT     (N_EDGES + N_NODES)   // with self loops = 850000
#define N_GRAPHS  64
#define HEADS     8
#define HID       16
#define DFEAT     128                   // HEADS*HID
#define N_CLASSES 10
#define N_LAYERS  5
#define NEG_SLOPE 0.2f

typedef __attribute__((ext_vector_type(16))) _Float16 v16h;
typedef __attribute__((ext_vector_type(8)))  _Float16 v8h;
typedef __attribute__((ext_vector_type(8)))  float    v8f;

// ---------------- helpers ----------------
__device__ __forceinline__ void atomicMaxFloat(float* addr, float val) {
    // classic signed/unsigned one-shot float max (location init'd to -FLT_MAX)
    if (val >= 0.0f) atomicMax((int*)addr, __float_as_int(val));
    else             atomicMin((unsigned int*)addr, __float_as_uint(val));
}

__device__ __forceinline__ void edge_src_dst(const int* __restrict__ ei, int e,
                                             int& s, int& d) {
    if (e < N_EDGES) { s = ei[e]; d = ei[N_EDGES + e]; }
    else             { s = d = e - N_EDGES; }           // appended self loops
}

// ---------------- fill ----------------
__global__ void fillf(float* __restrict__ p, int n, float v) {
    int i = blockIdx.x * 256 + threadIdx.x;
    if (i < n) p[i] = v;
}

// ---------------- weight convert: f32 W[l][k][n] -> f16 Wt[l][sel][n][k] ----------------
__global__ void wt_convert(const float* __restrict__ Wl, const float* __restrict__ Wr,
                           _Float16* __restrict__ Wt) {
    int i = blockIdx.x * 256 + threadIdx.x;           // over 5*2*128*128
    if (i >= N_LAYERS * 2 * DFEAT * DFEAT) return;
    int k   = i & (DFEAT - 1);
    int n   = (i >> 7) & (DFEAT - 1);
    int sel = (i >> 14) & 1;
    int l   = i >> 15;
    const float* W = sel ? Wr : Wl;
    Wt[i] = (_Float16)W[(size_t)l * DFEAT * DFEAT + (size_t)k * DFEAT + n];
}

// ---------------- WMMA GEMM: OUT = X @ W + b (both sides via blockIdx.y) ----------------
// grid: (3125, 2), block: 256 (8 waves, one 16x16 N-tile each). EXEC all-ones.
__global__ __launch_bounds__(256) void gat_gemm(
    const float* __restrict__ X,          // [N_NODES,128] f32
    const _Float16* __restrict__ Wt,      // [2][128][128] f16, n-major/k-contig
    const float* __restrict__ bl, const float* __restrict__ br,
    float* __restrict__ XL, float* __restrict__ XR)
{
    const int lane = threadIdx.x & 31;
    const int wave = threadIdx.x >> 5;       // 0..7 -> N tile
    const int sel  = blockIdx.y;             // 0: Wl->XL, 1: Wr->XR
    const int m0   = blockIdx.x * 16;
    const int n0   = wave * 16;
    const int hs   = lane >> 4;              // half-wave selector

    const _Float16* W    = Wt + (size_t)sel * DFEAT * DFEAT;
    const float*    bias = sel ? br : bl;
    float*          OUT  = sel ? XR : XL;

    const int mrow = m0 + (lane & 15);
    const int ncol = n0 + (lane & 15);
    const float*    arow = X + (size_t)mrow * DFEAT;
    const _Float16* wrow = W + (size_t)ncol * DFEAT;

    v8f acc = {};
    #pragma unroll
    for (int kb = 0; kb < 4; ++kb) {
        const int K0 = kb * 32;
        // ---- A fragment: a[j] = X[mrow][K0 + 16*(j>>3) + 8*hs + (j&7)] ----
        const int ka = K0 + 8 * hs;
        float4 f0 = ((const float4*)(arow + ka))[0];
        float4 f1 = ((const float4*)(arow + ka))[1];
        float4 f2 = ((const float4*)(arow + ka + 16))[0];
        float4 f3 = ((const float4*)(arow + ka + 16))[1];
        v16h a;
        a[0]=(_Float16)f0.x;  a[1]=(_Float16)f0.y;  a[2]=(_Float16)f0.z;  a[3]=(_Float16)f0.w;
        a[4]=(_Float16)f1.x;  a[5]=(_Float16)f1.y;  a[6]=(_Float16)f1.z;  a[7]=(_Float16)f1.w;
        a[8]=(_Float16)f2.x;  a[9]=(_Float16)f2.y;  a[10]=(_Float16)f2.z; a[11]=(_Float16)f2.w;
        a[12]=(_Float16)f3.x; a[13]=(_Float16)f3.y; a[14]=(_Float16)f3.z; a[15]=(_Float16)f3.w;
        // ---- B fragment: b[j] = W[K0 + 16*(j>>3) + 8*hs + (j&7)][ncol] (pre-transposed) ----
        const v8h* pb = (const v8h*)(wrow + K0 + 8 * hs);
        v8h b0 = pb[0], b1 = pb[2];
        v16h b;
        #pragma unroll
        for (int j = 0; j < 8; ++j) { b[j] = b0[j]; b[j + 8] = b1[j]; }

        acc = __builtin_amdgcn_wmma_f32_16x16x32_f16(
            false, a, false, b, (short)0, acc, false, false);
    }

    const float bval = bias[ncol];
    #pragma unroll
    for (int r = 0; r < 8; ++r) {
        const int M = m0 + r + 8 * hs;           // C/D layout: VGPR r -> row r (+8 upper half)
        OUT[(size_t)M * DFEAT + ncol] = acc[r] + bval;
    }
}

// ---------------- pass 1: logits + segment max ----------------
__global__ void edge_logits(const int* __restrict__ ei,
                            const float* __restrict__ XL, const float* __restrict__ XR,
                            const float* __restrict__ att,   // [8][16] this layer
                            float* __restrict__ logits, float* __restrict__ mbuf) {
    int idx = blockIdx.x * 256 + threadIdx.x;
    if (idx >= E_TOT * HEADS) return;
    int e = idx >> 3, h = idx & 7;
    int s, d; edge_src_dst(ei, e, s, d);
    const float* pl = XL + (size_t)s * DFEAT + h * HID;
    const float* pr = XR + (size_t)d * DFEAT + h * HID;
    const float* pa = att + h * HID;
    float acc = 0.0f;
    #pragma unroll
    for (int c = 0; c < HID; ++c) {
        float v = pl[c] + pr[c];
        v = v > 0.0f ? v : NEG_SLOPE * v;        // leaky_relu
        acc += pa[c] * v;
    }
    logits[idx] = acc;
    atomicMaxFloat(&mbuf[(size_t)d * HEADS + h], acc);
}

// ---------------- pass 2: p = exp(logit - m), segment sum (in-place over logits) ----------------
__global__ void edge_exp(const int* __restrict__ ei,
                         float* __restrict__ logits,
                         const float* __restrict__ mbuf, float* __restrict__ denom) {
    int idx = blockIdx.x * 256 + threadIdx.x;
    if (idx >= E_TOT * HEADS) return;
    int e = idx >> 3, h = idx & 7;
    int s, d; edge_src_dst(ei, e, s, d);
    float p = expf(logits[idx] - mbuf[(size_t)d * HEADS + h]);
    logits[idx] = p;
    atomicAdd(&denom[(size_t)d * HEADS + h], p);
}

// ---------------- pass 3: out[dst] += alpha * xl[src] ----------------
__global__ void edge_aggregate(const int* __restrict__ ei,
                               const float* __restrict__ XL,
                               const float* __restrict__ p, const float* __restrict__ denom,
                               float* __restrict__ agg) {
    int idx = blockIdx.x * 256 + threadIdx.x;    // E_TOT*128 threads exactly
    int e = idx >> 7, ch = idx & 127, h = ch >> 4;
    int s, d; edge_src_dst(ei, e, s, d);
    float alpha = p[(size_t)e * HEADS + h] / denom[(size_t)d * HEADS + h];
    atomicAdd(&agg[(size_t)d * DFEAT + ch], XL[(size_t)s * DFEAT + ch] * alpha);
}

// ---------------- bias + ELU into hcur ----------------
__global__ void finish_elu(const float* __restrict__ agg, const float* __restrict__ bias,
                           float* __restrict__ hcur) {
    int idx = blockIdx.x * 256 + threadIdx.x;    // N_NODES*128 exactly
    float v = agg[idx] + bias[idx & 127];
    hcur[idx] = v > 0.0f ? v : (expf(v) - 1.0f); // elu
}

// ---------------- global mean pool (atomic accumulate) ----------------
__global__ void pool_sum(const float* __restrict__ h, const int* __restrict__ batch,
                         float* __restrict__ pooled, float* __restrict__ counts) {
    int idx = blockIdx.x * 256 + threadIdx.x;    // N_NODES*128 exactly
    int n = idx >> 7, ch = idx & 127;
    int g = batch[n];
    atomicAdd(&pooled[(size_t)g * DFEAT + ch], h[idx]);
    if (ch == 0) atomicAdd(&counts[g], 1.0f);
}

// ---------------- head: mean, linear, log_softmax ----------------
__global__ void final_head(const float* __restrict__ pooled, const float* __restrict__ counts,
                           const float* __restrict__ Wout, const float* __restrict__ bout,
                           float* __restrict__ out) {
    int g = threadIdx.x;
    if (g >= N_GRAPHS) return;
    float cnt = counts[g]; cnt = cnt > 1.0f ? cnt : 1.0f;
    float inv = 1.0f / cnt;
    float lg[N_CLASSES];
    #pragma unroll
    for (int c = 0; c < N_CLASSES; ++c) lg[c] = bout[c];
    for (int d = 0; d < DFEAT; ++d) {
        float v = pooled[(size_t)g * DFEAT + d] * inv;
        #pragma unroll
        for (int c = 0; c < N_CLASSES; ++c) lg[c] += v * Wout[d * N_CLASSES + c];
    }
    float mx = lg[0];
    #pragma unroll
    for (int c = 1; c < N_CLASSES; ++c) mx = fmaxf(mx, lg[c]);
    float se = 0.0f;
    #pragma unroll
    for (int c = 0; c < N_CLASSES; ++c) se += expf(lg[c] - mx);
    float lse = mx + logf(se);
    #pragma unroll
    for (int c = 0; c < N_CLASSES; ++c) out[g * N_CLASSES + c] = lg[c] - lse;
}

// ---------------- launcher ----------------
extern "C" void kernel_launch(void* const* d_in, const int* in_sizes, int n_in,
                              void* d_out, int out_size, void* d_ws, size_t ws_size,
                              hipStream_t stream) {
    const float* x     = (const float*)d_in[0];
    const int*   ei    = (const int*)  d_in[1];
    const int*   batch = (const int*)  d_in[2];
    const float* Wl    = (const float*)d_in[3];
    const float* bl    = (const float*)d_in[4];
    const float* Wr    = (const float*)d_in[5];
    const float* br    = (const float*)d_in[6];
    const float* att   = (const float*)d_in[7];
    const float* bias  = (const float*)d_in[8];
    const float* Wout  = (const float*)d_in[9];
    const float* bout  = (const float*)d_in[10];
    float* out = (float*)d_out;

    char* ws = (char*)d_ws;
    size_t off = 0;
    auto take = [&](size_t bytes) -> char* {
        char* p = ws + off; off += (bytes + 255) & ~(size_t)255; return p;
    };
    float*     hcur   = (float*)take((size_t)N_NODES * DFEAT * 4);
    float*     xl     = (float*)take((size_t)N_NODES * DFEAT * 4);
    float*     xr     = (float*)take((size_t)N_NODES * DFEAT * 4);
    float*     agg    = (float*)take((size_t)N_NODES * DFEAT * 4);
    float*     logits = (float*)take((size_t)E_TOT * HEADS * 4);
    float*     mbuf   = (float*)take((size_t)N_NODES * HEADS * 4);
    float*     denom  = (float*)take((size_t)N_NODES * HEADS * 4);
    _Float16*  wt     = (_Float16*)take((size_t)N_LAYERS * 2 * DFEAT * DFEAT * 2);
    float*     pooled = (float*)take((size_t)N_GRAPHS * DFEAT * 4);
    float*     counts = (float*)take((size_t)N_GRAPHS * 4);

    const int NB_NH  = (N_NODES * HEADS + 255) / 256;          // 1563
    const int NB_ND  = (N_NODES * DFEAT) / 256;                // 25000 (exact)
    const int NB_EH  = (E_TOT * HEADS + 255) / 256;            // 26563
    const int NB_ED  = (E_TOT * DFEAT) / 256;                  // 425000 (exact)
    const int NB_WT  = (N_LAYERS * 2 * DFEAT * DFEAT) / 256;   // 640 (exact)

    // one-time weight transpose + f16 convert
    wt_convert<<<NB_WT, 256, 0, stream>>>(Wl, Wr, wt);

    for (int l = 0; l < N_LAYERS; ++l) {
        const float* Xin = (l == 0) ? x : hcur;
        gat_gemm<<<dim3(N_NODES / 16, 2), 256, 0, stream>>>(
            Xin, wt + (size_t)l * 2 * DFEAT * DFEAT,
            bl + l * DFEAT, br + l * DFEAT, xl, xr);

        fillf<<<NB_NH, 256, 0, stream>>>(mbuf,  N_NODES * HEADS, -3.402823466e38f);
        fillf<<<NB_NH, 256, 0, stream>>>(denom, N_NODES * HEADS, 0.0f);
        fillf<<<NB_ND, 256, 0, stream>>>(agg,   N_NODES * DFEAT, 0.0f);

        edge_logits<<<NB_EH, 256, 0, stream>>>(ei, xl, xr, att + l * HEADS * HID,
                                               logits, mbuf);
        edge_exp<<<NB_EH, 256, 0, stream>>>(ei, logits, mbuf, denom);
        edge_aggregate<<<NB_ED, 256, 0, stream>>>(ei, xl, logits, denom, agg);
        finish_elu<<<NB_ND, 256, 0, stream>>>(agg, bias + l * DFEAT, hcur);
    }

    fillf<<<(N_GRAPHS * DFEAT + 255) / 256, 256, 0, stream>>>(pooled, N_GRAPHS * DFEAT, 0.0f);
    fillf<<<1, 256, 0, stream>>>(counts, N_GRAPHS, 0.0f);
    pool_sum<<<NB_ND, 256, 0, stream>>>(hcur, batch, pooled, counts);
    final_head<<<1, 64, 0, stream>>>(pooled, counts, Wout, bout, out);
}
